// RealisticBlock_76089640616227
// MI455X (gfx1250) — matrix operand order repaired and checked by reading, not compiled, not run on hardware
//
#include <hip/hip_runtime.h>

typedef __attribute__((ext_vector_type(16))) _Float16 v16h;
typedef __attribute__((ext_vector_type(8)))  _Float16 v8h;
typedef __attribute__((ext_vector_type(8)))  float    v8f;

#define D     128
#define E     8
#define I     16
#define NTOK  65536      // B*T = 8*8192
#define WAVES 4
#define TPB   (32*WAVES)
#define TILE  16         // tokens per wave (WMMA M)

// LDS strides (elements), padded to dodge bank conflicts, 16B-aligned rows
#define W1_STR 136   // halfs  : w1T  [E][I=16][D=128]
#define W2_STR 24    // halfs  : w2T  [E][D=128][I=16]  (+1 zero row)
#define W2_ZROW (E*D)  // index of the zero row lanes 16-31 read (K=16..31 pad)
#define XN_STR 136   // halfs  : x_norm f16 [16][128]
#define X_STR  132   // floats : x f32     [16][128]
#define Y_STRF 20    // floats : gelu(y)   [16][16]

__global__ __launch_bounds__(TPB) void moe_block_kernel(
    const float* __restrict__ x,      const float* __restrict__ ln_g,
    const float* __restrict__ ln_b,   const float* __restrict__ gate_w,
    const float* __restrict__ gate_b, const float* __restrict__ w1,
    const float* __restrict__ w2,
    float* __restrict__ out0, float* __restrict__ out_logits,
    float* __restrict__ out_eo)
{
  __shared__ __align__(16) _Float16 s_w1[E*I*W1_STR];
  __shared__ __align__(16) _Float16 s_w2[(E*D + 1)*W2_STR];
  __shared__ __align__(16) float    s_gw[D*E];
  __shared__ float s_g[D];
  __shared__ float s_b[D];
  __shared__ __align__(16) float    s_x [WAVES][TILE*X_STR];
  __shared__ __align__(16) _Float16 s_xn[WAVES][TILE*XN_STR];
  __shared__ __align__(16) float    s_y [WAVES][TILE*Y_STRF];
  __shared__ float s_rw[WAVES][TILE*E];

  const int tid = threadIdx.x;

  // ---- stage weights once per block (f32 -> f16, transposed for WMMA B) ----
  for (int idx = tid; idx < E*D*I; idx += TPB) {          // w1 [E][D][I]
    int e = idx >> 11, rem = idx & 2047, c = rem >> 4, i = rem & 15;
    s_w1[(e*I + i)*W1_STR + c] = (_Float16)w1[idx];
  }
  for (int idx = tid; idx < E*I*D; idx += TPB) {          // w2 [E][I][D]
    int e = idx >> 11, rem = idx & 2047, i = rem >> 7, c = rem & 127;
    s_w2[(e*D + c)*W2_STR + i] = (_Float16)w2[idx];
  }
  for (int idx = tid; idx < W2_STR; idx += TPB)           // zero pad row (K=16..31)
    s_w2[W2_ZROW*W2_STR + idx] = (_Float16)0.f;
  for (int idx = tid; idx < D*E; idx += TPB) s_gw[idx] = gate_w[idx];
  for (int idx = tid; idx < D;   idx += TPB) { s_g[idx] = ln_g[idx]; s_b[idx] = ln_b[idx]; }
  __syncthreads();

  const int wave   = tid >> 5;
  const int lane   = tid & 31;
  const int laneHi = lane >> 4;     // 0: lanes 0-15, 1: lanes 16-31
  const int lm     = lane & 15;
  const long tile0 = (long)(blockIdx.x * WAVES + wave) * TILE;

  float*    sx  = s_x[wave];
  _Float16* sxn = s_xn[wave];
  float*    sy  = s_y[wave];
  float*    srw = s_rw[wave];

  // ---- 1: load 16x128 f32 token tile (one row per iteration, coalesced) ----
  {
    const float4* xg = reinterpret_cast<const float4*>(x + tile0 * D);
#pragma unroll
    for (int r = 0; r < TILE; ++r) {
      float4 v = xg[r*(D/4) + lane];
      *reinterpret_cast<float4*>(sx + r*X_STR + lane*4) = v;
    }
  }
  __builtin_amdgcn_wave_barrier();

  // ---- 2: LayerNorm stats: 2 lanes per token, interleaved columns ----
  const int t  = lane >> 1;
  const int hh = lane & 1;
  float sum = 0.f, sq = 0.f;
  {
    const float* xr = sx + t*X_STR;
#pragma unroll 8
    for (int j = 0; j < 64; ++j) { float v = xr[2*j + hh]; sum += v; sq += v*v; }
  }
  sum += __shfl_xor(sum, 1, 32);
  sq  += __shfl_xor(sq,  1, 32);
  const float mu   = sum * (1.f/128.f);
  const float rstd = rsqrtf(sq*(1.f/128.f) - mu*mu + 1e-5f);

  // ---- 3: normalize -> f16 LDS (packed 32b stores) + gating logits ----
  float acc[E];
#pragma unroll
  for (int e = 0; e < E; ++e) acc[e] = 0.f;
  {
    const int c0 = hh * 64;
    const float2* xr2 = reinterpret_cast<const float2*>(sx + t*X_STR + c0);
#pragma unroll 4
    for (int j = 0; j < 32; ++j) {
      float2 v2 = xr2[j];
      int c = c0 + 2*j;
      float v0 = (v2.x - mu)*rstd*s_g[c]   + s_b[c];
      float v1 = (v2.y - mu)*rstd*s_g[c+1] + s_b[c+1];
#pragma unroll
      for (int e = 0; e < E; ++e)
        acc[e] += v0*s_gw[c*E+e] + v1*s_gw[(c+1)*E+e];
      union { _Float16 h[2]; unsigned u; } p;
      p.h[0] = (_Float16)v0; p.h[1] = (_Float16)v1;
      *reinterpret_cast<unsigned*>(sxn + t*XN_STR + c) = p.u;
    }
  }
  float logitv[E], rwv[E];
  float mx = -1e30f;
#pragma unroll
  for (int e = 0; e < E; ++e) {
    acc[e] += __shfl_xor(acc[e], 1, 32);
    logitv[e] = acc[e] + gate_b[e];
    mx = fmaxf(mx, logitv[e]);
  }
  float se = 0.f;
#pragma unroll
  for (int e = 0; e < E; ++e) { rwv[e] = expf(logitv[e] - mx); se += rwv[e]; }
  const float inv = 1.f / se;
  if (hh == 0) {
#pragma unroll
    for (int e = 0; e < E; ++e) {
      srw[t*E + e] = rwv[e]*inv;
      __builtin_nontemporal_store(logitv[e], &out_logits[(tile0 + t)*E + e]);
    }
  }
  __builtin_amdgcn_wave_barrier();

  // ---- 4: A fragments (16x32 f16 per chunk, ISA VGPR layout) ----
  // lane holds row M=lm; halfs 0..7 -> K = 8*laneHi+0..7, halfs 8..15 -> K = 16+8*laneHi+0..7
  v16h a[4];
  {
    const _Float16* xb = sxn + lm*XN_STR + 8*laneHi;
#pragma unroll
    for (int c = 0; c < 4; ++c) {
      v8h lo = *reinterpret_cast<const v8h*>(xb + c*32);
      v8h hi = *reinterpret_cast<const v8h*>(xb + c*32 + 16);
      a[c] = __builtin_shufflevector(lo, hi, 0,1,2,3,4,5,6,7,8,9,10,11,12,13,14,15);
    }
  }

  v8f agg[8];
#pragma unroll
  for (int tt = 0; tt < 8; ++tt) { v8f z = {}; agg[tt] = z; }

  // ---- 5: expert loop ----
  for (int e = 0; e < E; ++e) {
    // GEMM1: y[16x16] = x_norm[16x128] * w1[e][128x16], f32 accumulate
    v8f yv = {};
    const _Float16* w1b = s_w1 + (e*I + lm)*W1_STR + 8*laneHi;    // B col n=lm, K = 8*laneHi+0..7, 16+8*laneHi+0..7
#pragma unroll
    for (int c = 0; c < 4; ++c) {
      v8h lo = *reinterpret_cast<const v8h*>(w1b + c*32);
      v8h hi = *reinterpret_cast<const v8h*>(w1b + c*32 + 16);
      v16h bf = __builtin_shufflevector(lo, hi, 0,1,2,3,4,5,6,7,8,9,10,11,12,13,14,15);
      yv = __builtin_amdgcn_wmma_f32_16x16x32_f16(false, a[c], false, bf,
                                                  (short)0, yv, false, false);
    }
    // exact-erf GELU, stage to LDS (C-layout -> row-major f32)
#pragma unroll
    for (int r = 0; r < 8; ++r) {
      float v = yv[r];
      float g = 0.5f * v * (1.0f + erff(v * 0.70710678118654752440f));
      sy[(r + 8*laneHi)*Y_STRF + lm] = g;
    }
    __builtin_amdgcn_wave_barrier();

    // A2: gelu(y) as 16x32 f16 with K=16..31 zero-padded
    v16h a2;
    {
      const float* yr = sy + lm*Y_STRF;
      float4 f0 = *reinterpret_cast<const float4*>(yr);
      float4 f1 = *reinterpret_cast<const float4*>(yr + 4);
      a2[0]=(_Float16)f0.x; a2[1]=(_Float16)f0.y; a2[2]=(_Float16)f0.z; a2[3]=(_Float16)f0.w;
      a2[4]=(_Float16)f1.x; a2[5]=(_Float16)f1.y; a2[6]=(_Float16)f1.z; a2[7]=(_Float16)f1.w;
#pragma unroll
      for (int k = 8; k < 16; ++k) a2[k] = laneHi ? (_Float16)0.f : (_Float16)yr[k];
    }
    __builtin_amdgcn_wave_barrier();

    float rw[8];
#pragma unroll
    for (int r = 0; r < 8; ++r) rw[r] = srw[(r + 8*laneHi)*E + e];

    float* eo_base = out_eo + ((tile0 + 8*laneHi)*E + e)*D + lm;

    // GEMM2: out[16x128] = a2[16x16] * w2[e][16x128], 8 column tiles.
    // Lanes 16-31 (K=16..31) read the dedicated zero row -> no per-element cndmask.
#pragma unroll
    for (int tt = 0; tt < 8; ++tt) {
      const int row = laneHi ? W2_ZROW : (e*D + tt*16 + lm);
      const _Float16* wb = s_w2 + row*W2_STR;
      v8h lo = *reinterpret_cast<const v8h*>(wb);
      v8h hi = *reinterpret_cast<const v8h*>(wb + 8);
      v16h b2 = __builtin_shufflevector(lo, hi, 0,1,2,3,4,5,6,7,8,9,10,11,12,13,14,15);
      v8f o = {};
      o = __builtin_amdgcn_wmma_f32_16x16x32_f16(false, a2, false, b2,
                                                 (short)0, o, false, false);
#pragma unroll
      for (int r = 0; r < 8; ++r) {
        __builtin_nontemporal_store(o[r], &eo_base[(long)r*(E*D) + tt*16]); // moe_experts_out
        agg[tt][r] += rw[r] * o[r];                                         // routing-weighted sum
      }
    }
  }

  // ---- 6: residual add + store final output ----
  {
    float* o0 = out0 + (tile0 + 8*laneHi)*D + lm;
#pragma unroll
    for (int tt = 0; tt < 8; ++tt)
#pragma unroll
      for (int r = 0; r < 8; ++r) {
        float xv = sx[(r + 8*laneHi)*X_STR + tt*16 + lm];
        __builtin_nontemporal_store(xv + agg[tt][r], &o0[(long)r*D + tt*16]);
      }
  }
}

extern "C" void kernel_launch(void* const* d_in, const int* in_sizes, int n_in,
                              void* d_out, int out_size, void* d_ws, size_t ws_size,
                              hipStream_t stream) {
  const float* x      = (const float*)d_in[0];
  const float* ln_g   = (const float*)d_in[1];
  const float* ln_b   = (const float*)d_in[2];
  const float* gate_w = (const float*)d_in[3];
  const float* gate_b = (const float*)d_in[4];
  const float* w1     = (const float*)d_in[5];
  const float* w2     = (const float*)d_in[6];
  float* out        = (float*)d_out;
  float* out0       = out;                                   // [N,128]
  float* out_logits = out + (long)NTOK*D;                    // [N,8]
  float* out_eo     = out + (long)NTOK*D + (long)NTOK*E;     // [N,8,128]
  dim3 grid(NTOK / (WAVES*TILE));   // 1024 blocks x 4 waves x 16 tokens
  moe_block_kernel<<<grid, TPB, 0, stream>>>(x, ln_g, ln_b, gate_w, gate_b,
                                             w1, w2, out0, out_logits, out_eo);
}
